// MultiheadGraphAttention_14800457302611
// MI455X (gfx1250) — matrix operand order repaired and checked by reading, not compiled, not run on hardware
//
#include <hip/hip_runtime.h>
#include <math.h>

// Problem constants (match reference)
#define B_ 2
#define N_ 2048
#define E_ 256
#define H_ 4
#define DK_ 64

typedef __attribute__((ext_vector_type(16))) __bf16 v16bf;
typedef __attribute__((ext_vector_type(8)))  float  v8f;
typedef unsigned __attribute__((ext_vector_type(4))) v4u;
typedef int      __attribute__((ext_vector_type(4))) v4i;
typedef int      __attribute__((ext_vector_type(8))) v8i;

// Tensor Data Mover availability (arity differs across toolchains)
#if defined(__has_builtin)
#if __has_builtin(__builtin_amdgcn_tensor_load_to_lds) && \
    __has_builtin(__builtin_amdgcn_s_wait_tensorcnt)
#define USE_TDM 1
#endif
#endif
#ifndef USE_TDM
#define USE_TDM 0
#endif

// ---------------------------------------------------------------------------
// WMMA helpers (CDNA5 gfx1250, wave32). Layouts used here (A/B k order is this code's own, not the ISA table's):
//  A (16x32 bf16): lane L -> row = L&15; elem j = K[(L<16?0:16)+j]
//                  (same k order as B)
//  B (32x16 bf16): lane L -> col = L&15; elem j = K[(L<16?0:16)+j]
//  C/D (16x16 f32): lane L -> col = L&15; vgpr r -> row = (L<16?0:8)+r
// All fragment loaders are contiguous per lane -> lower to *_b128.
// ---------------------------------------------------------------------------
__device__ inline v8f wmma_bf16(v16bf a, v16bf b, v8f c) {
  return __builtin_amdgcn_wmma_f32_16x16x32_bf16(
      /*neg_a=*/false, a, /*neg_b=*/false, b,
      /*c_mod=*/(short)0, c, /*reuse_a=*/false, /*reuse_b=*/false);
}

__device__ inline v16bf load_a_frag(const __bf16* src, int ld, int lane) {
  const int hl = lane >> 4, ln = lane & 15;
  const __bf16* p = src + ln * ld + hl * 16;
  v16bf a;
#pragma unroll
  for (int j = 0; j < 8; ++j) a[j] = p[j];
#pragma unroll
  for (int j = 0; j < 8; ++j) a[8 + j] = p[8 + j];
  return a;
}

// B-fragment where memory holds B TRANSPOSED: src row-major (N x K),
// B[k][n] = src[n][k]. src points at (n0, k0). Contiguous per lane.
__device__ inline v16bf load_bT_frag(const __bf16* src, int ld, int lane) {
  const int hl = lane >> 4, ln = lane & 15;
  const __bf16* p = src + ln * ld + hl * 16;
  v16bf b;
#pragma unroll
  for (int j = 0; j < 16; ++j) b[j] = p[j];
  return b;
}

#if USE_TDM
// ---------------------------------------------------------------------------
// Issue one TDM 2-D tile load: global (td1 rows x td0 cols, row stride
// stride0, 2-byte elems) -> LDS (tile1 x tile0, packed). D# layout per
// CDNA5 ISA ch.8: group0 {count=1, lds_addr, global_addr[56:0], type=2},
// group1 {data_size=2B, tensor dims, tile dims, dim0 stride}.
// Issued per-wave (EXEC ignored); completion tracked with TENSORcnt.
// ---------------------------------------------------------------------------
__device__ inline void tdm_load_2d(const void* gaddr, const void* lds,
                                   unsigned td0, unsigned td1,
                                   unsigned tile0, unsigned tile1,
                                   unsigned stride0) {
  unsigned long long ga = (unsigned long long)(size_t)gaddr;
  unsigned la = (unsigned)(size_t)lds;  // LDS byte address (low 32 bits)
  v4u g0;
  g0[0] = 1u;                                                  // count=1
  g0[1] = la;                                                  // lds_addr
  g0[2] = (unsigned)ga;                                        // gaddr[31:0]
  g0[3] = (unsigned)((ga >> 32) & 0x01FFFFFFu) | (2u << 30);   // gaddr[56:32]|type=2
  v8i g1;
  g1[0] = (int)(1u << 16);                                     // data_size=1 (2B)
  g1[1] = (int)((td0 & 0xFFFFu) << 16);                        // tensor_dim0 lo
  g1[2] = (int)(((td0 >> 16) & 0xFFFFu) | ((td1 & 0xFFFFu) << 16));
  g1[3] = (int)(((td1 >> 16) & 0xFFFFu) | (tile0 << 16));      // tile_dim0
  g1[4] = (int)tile1;                                          // tile_dim1, dim2=0
  g1[5] = (int)stride0;                                        // dim0 stride lo
  g1[6] = 0;
  g1[7] = 0;
  v4i z4 = {0, 0, 0, 0};
#if __clang_major__ >= 23
  v8i z8 = {0, 0, 0, 0, 0, 0, 0, 0};
  __builtin_amdgcn_tensor_load_to_lds(g0, g1, z4, z4, z8, 0);
#else
  __builtin_amdgcn_tensor_load_to_lds(g0, g1, z4, z4, 0);
#endif
}
#endif  // USE_TDM

// ---------------------------------------------------------------------------
// Kernel 0: one-time weight prep.
//  Wq/Wk/Wv (H,E,DK) fp32 -> WT (H,DK,E) bf16 ; Wo (E,E) fp32 -> bf16
// ---------------------------------------------------------------------------
__global__ void prep_weights_kernel(const float* __restrict__ Wq,
                                    const float* __restrict__ Wk,
                                    const float* __restrict__ Wv,
                                    const float* __restrict__ Wo,
                                    __bf16* __restrict__ WqT, __bf16* __restrict__ WkT,
                                    __bf16* __restrict__ WvT, __bf16* __restrict__ Wob) {
  int t = blockIdx.x * blockDim.x + threadIdx.x;  // 4 * 65536 threads
  int mat = t >> 16;
  int i = t & 0xFFFF;
  if (mat < 3) {
    const float* W = (mat == 0) ? Wq : (mat == 1) ? Wk : Wv;
    __bf16* WT = (mat == 0) ? WqT : (mat == 1) ? WkT : WvT;
    int k = i & (E_ - 1);
    int n = (i >> 8) & (DK_ - 1);
    int h = i >> 14;
    WT[i] = (__bf16)W[((size_t)h * E_ + k) * DK_ + n];
  } else {
    Wob[i] = (__bf16)Wo[i];
  }
}

// ---------------------------------------------------------------------------
// Kernel 1: adjacency int32 (B,N,N) -> bitmask u32 (B,N,N/32).
// ---------------------------------------------------------------------------
__global__ void adj_to_bits_kernel(const int* __restrict__ adj,
                                   unsigned* __restrict__ bits, int total_words) {
  int w = blockIdx.x * blockDim.x + threadIdx.x;
  if (w >= total_words) return;
  const int* p = adj + (size_t)w * 32;
  unsigned m = 0;
#pragma unroll
  for (int i = 0; i < 32; ++i) m |= (p[i] > 0 ? 1u : 0u) << i;
  bits[w] = m;
}

// ---------------------------------------------------------------------------
// Kernel 2: fused QKV projection (128 rows x one head). Q,K -> (B,H,N,DK)
// bf16; V -> (B,H,DK,N) bf16 transposed (one b128 store per column tile).
// ---------------------------------------------------------------------------
__global__ void qkv_kernel(const float* __restrict__ X,
                           const __bf16* __restrict__ WqT,
                           const __bf16* __restrict__ WkT,
                           const __bf16* __restrict__ WvT,
                           __bf16* __restrict__ Qo, __bf16* __restrict__ Ko,
                           __bf16* __restrict__ Vt) {
  const int wg = blockIdx.x;
  const int h  = wg % H_;
  const int rt = (wg / H_) % (N_ / 128);
  const int b  = wg / (H_ * (N_ / 128));
  const int tid = threadIdx.x;
  const int wave = tid >> 5, lane = tid & 31;
  const int hl = lane >> 4, ln = lane & 15;

  __shared__ __bf16 Xs[128 * 32];
  __shared__ __bf16 WsT[3][64 * 32];

  v8f acc[3][4] = {};
  const int row0 = rt * 128;
  const __bf16* WTs[3] = {WqT, WkT, WvT};

  for (int k0 = 0; k0 < E_; k0 += 32) {
    __syncthreads();
    {
      const int r = tid >> 1, c = (tid & 1) * 16;
      const float4* xs = (const float4*)(X + ((size_t)b * N_ + row0 + r) * E_ + k0 + c);
      union { __bf16 h16[16]; uint4 q[2]; } u;
#pragma unroll
      for (int v = 0; v < 4; ++v) {
        float4 f = xs[v];
        u.h16[v * 4 + 0] = (__bf16)f.x; u.h16[v * 4 + 1] = (__bf16)f.y;
        u.h16[v * 4 + 2] = (__bf16)f.z; u.h16[v * 4 + 3] = (__bf16)f.w;
      }
      uint4* dst = (uint4*)&Xs[r * 32 + c];
      dst[0] = u.q[0]; dst[1] = u.q[1];
    }
    {
      const int n = tid >> 2, seg = (tid & 3) * 8;
#pragma unroll
      for (int m = 0; m < 3; ++m)
        *(uint4*)&WsT[m][n * 32 + seg] =
            *(const uint4*)(WTs[m] + ((size_t)h * DK_ + n) * E_ + k0 + seg);
    }
    __syncthreads();

    v16bf a = load_a_frag(&Xs[wave * 16 * 32], 32, lane);
#pragma unroll
    for (int m = 0; m < 3; ++m)
#pragma unroll
      for (int ct = 0; ct < 4; ++ct)
        acc[m][ct] = wmma_bf16(a, load_bT_frag(&WsT[m][ct * 16 * 32], 32, lane), acc[m][ct]);
  }

  const size_t bh = (size_t)b * H_ + h;
  const int rbase = row0 + wave * 16 + hl * 8;
#pragma unroll
  for (int ct = 0; ct < 4; ++ct)
#pragma unroll
    for (int r = 0; r < 8; ++r) {
      size_t o = (bh * N_ + rbase + r) * DK_ + ct * 16 + ln;
      Qo[o] = (__bf16)acc[0][ct][r];
      Ko[o] = (__bf16)acc[1][ct][r];
    }
#pragma unroll
  for (int ct = 0; ct < 4; ++ct) {
    union { __bf16 h16[8]; uint4 q; } u;
#pragma unroll
    for (int r = 0; r < 8; ++r) u.h16[r] = (__bf16)acc[2][ct][r];
    *(uint4*)(Vt + (bh * DK_ + ct * 16 + ln) * N_ + rbase) = u.q;
  }
}

// ---------------------------------------------------------------------------
// Kernel 3: flash-style masked attention, 128 query rows per workgroup.
// TDM path: double-buffered K/V LDS tiles; wave 0 issues tensor_load_to_lds
// for tile t+1 while all waves compute on tile t; one s_wait_tensorcnt +
// barrier per iteration. Fallback: synchronous uint4 staging.
// ---------------------------------------------------------------------------
__global__ void attn_kernel(const __bf16* __restrict__ Q,
                            const __bf16* __restrict__ K,
                            const __bf16* __restrict__ Vt,
                            const unsigned* __restrict__ adjbits,
                            __bf16* __restrict__ heads) {
  const int bhid = blockIdx.x;
  const int b = bhid / H_, h = bhid % H_;
  const int rt = blockIdx.y;
  const int tid = threadIdx.x, wave = tid >> 5, lane = tid & 31;
  const int hl = lane >> 4, ln = lane & 15;
  const float NEG_INF = -__builtin_inff();
  const int TILES = N_ / 32;

  __shared__ __bf16 Ks[2][32 * 64];    // (key x d) row-major, double buffer
  __shared__ __bf16 Vs[2][64 * 32];    // (d x key) transposed, double buffer
  __shared__ __bf16 Ps[8][16 * 32];    // per-wave P transpose scratch

  const size_t bh = (size_t)b * H_ + h;
  const size_t qk_base = bh * (size_t)N_ * DK_;
  const size_t vt_base = bh * (size_t)DK_ * N_;
  const int qrow0 = rt * 128 + wave * 16;
  const int words = N_ / 32;

  v16bf qf0 = load_a_frag(Q + qk_base + (size_t)qrow0 * DK_ + 0,  DK_, lane);
  v16bf qf1 = load_a_frag(Q + qk_base + (size_t)qrow0 * DK_ + 32, DK_, lane);

  v8f o[4] = {};
  float m[8], l[8];
#pragma unroll
  for (int r = 0; r < 8; ++r) { m[r] = NEG_INF; l[r] = 0.0f; }
  const float scale = 0.125f;  // 1/sqrt(64)

#if USE_TDM
  int cur = 0;
  if (wave == 0) {  // prologue: DMA tile 0 into buffer 0
    tdm_load_2d(K + qk_base, Ks[0], DK_, N_, DK_, 32, DK_);
    tdm_load_2d(Vt + vt_base, Vs[0], N_, DK_, 32, DK_, N_);
    __builtin_amdgcn_s_wait_tensorcnt(0);
  }
  __syncthreads();
#endif

  for (int kt = 0; kt < TILES; ++kt) {
#if USE_TDM
    if (wave == 0 && kt + 1 < TILES) {  // async DMA of next tile, overlapped
      tdm_load_2d(K + qk_base + (size_t)(kt + 1) * 32 * DK_, Ks[cur ^ 1],
                  DK_, N_, DK_, 32, DK_);
      tdm_load_2d(Vt + vt_base + (size_t)(kt + 1) * 32, Vs[cur ^ 1],
                  N_, DK_, 32, DK_, N_);
    }
    const __bf16* Kst = Ks[cur];
    const __bf16* Vst = Vs[cur];
#else
    __syncthreads();
    {
      ((uint4*)Ks[0])[tid] = ((const uint4*)(K + qk_base + (size_t)kt * 32 * DK_))[tid];
      const int d = tid >> 2, seg = (tid & 3) * 8;
      *(uint4*)&Vs[0][d * 32 + seg] =
          *(const uint4*)(Vt + vt_base + (size_t)d * N_ + kt * 32 + seg);
    }
    if (kt + 1 < TILES)
      __builtin_prefetch(K + qk_base + (size_t)(kt + 1) * 32 * DK_, 0, 0);
    __syncthreads();
    const __bf16* Kst = Ks[0];
    const __bf16* Vst = Vs[0];
#endif

    // Scores S (16 rows x 32 keys) = Q . K^T, fp32 accum
    v8f s[2] = {};
#pragma unroll
    for (int nt = 0; nt < 2; ++nt) {
      s[nt] = wmma_bf16(qf0, load_bT_frag(Kst + nt * 16 * 64 + 0,  64, lane), s[nt]);
      s[nt] = wmma_bf16(qf1, load_bT_frag(Kst + nt * 16 * 64 + 32, 64, lane), s[nt]);
    }

    // Adjacency bit words: one u32 per query row covers this tile's 32 keys
    unsigned aw[8];
#pragma unroll
    for (int r = 0; r < 8; ++r)
      aw[r] = adjbits[((size_t)b * N_ + qrow0 + hl * 8 + r) * words + kt];

    float p0[8], p1[8], alpha[8];
#pragma unroll
    for (int r = 0; r < 8; ++r) {
      float s0 = ((aw[r] >> ln) & 1u)        ? s[0][r] * scale : NEG_INF;
      float s1 = ((aw[r] >> (16 + ln)) & 1u) ? s[1][r] * scale : NEG_INF;
      float mx = fmaxf(s0, s1);
#pragma unroll
      for (int off = 8; off; off >>= 1) mx = fmaxf(mx, __shfl_xor(mx, off, 16));
      float mnew = fmaxf(m[r], mx);
      float a_ = (mnew == NEG_INF) ? 1.0f : __expf(m[r] - mnew);
      float e0 = (s0 == NEG_INF) ? 0.0f : __expf(s0 - mnew);
      float e1 = (s1 == NEG_INF) ? 0.0f : __expf(s1 - mnew);
      float rs = e0 + e1;
#pragma unroll
      for (int off = 8; off; off >>= 1) rs += __shfl_xor(rs, off, 16);
      l[r] = l[r] * a_ + rs;
      m[r] = mnew;
      alpha[r] = a_;
      p0[r] = e0; p1[r] = e1;
    }

#pragma unroll
    for (int ct = 0; ct < 4; ++ct)
#pragma unroll
      for (int r = 0; r < 8; ++r) o[ct][r] *= alpha[r];

    // Flip P from C-layout to A-layout through per-wave LDS scratch
    __bf16* myP = Ps[wave];
#pragma unroll
    for (int r = 0; r < 8; ++r) {
      myP[(hl * 8 + r) * 32 + ln]      = (__bf16)p0[r];
      myP[(hl * 8 + r) * 32 + 16 + ln] = (__bf16)p1[r];
    }
    v16bf pf = load_a_frag(myP, 32, lane);  // DS pipeline is in-order per wave

    // O += P . V : Vs is (d x key) so B-fragments are contiguous b128 reads
#pragma unroll
    for (int ct = 0; ct < 4; ++ct)
      o[ct] = wmma_bf16(pf, load_bT_frag(Vst + ct * 16 * 32, 32, lane), o[ct]);

#if USE_TDM
    if (wave == 0 && kt + 1 < TILES) __builtin_amdgcn_s_wait_tensorcnt(0);
    __syncthreads();  // publish next buffer / retire reads of current one
    cur ^= 1;
#endif
  }

  float invl[8];
#pragma unroll
  for (int r = 0; r < 8; ++r) invl[r] = (l[r] > 0.0f) ? 1.0f / l[r] : 0.0f;
#pragma unroll
  for (int ct = 0; ct < 4; ++ct)
#pragma unroll
    for (int r = 0; r < 8; ++r) {
      int row = qrow0 + hl * 8 + r;
      heads[(((size_t)b * N_ + row) * H_ + h) * DK_ + ct * 16 + ln] =
          (__bf16)(o[ct][r] * invl[r]);
    }
}

// ---------------------------------------------------------------------------
// Kernel 4: output projection  out = concat @ Wo^T + bo  (fp32 out).
// ---------------------------------------------------------------------------
__global__ void proj_kernel(const __bf16* __restrict__ heads,
                            const __bf16* __restrict__ Wob,
                            const float* __restrict__ bo,
                            float* __restrict__ out) {
  const int rt = blockIdx.x;
  const int c0 = blockIdx.y * 64;
  const int tid = threadIdx.x, wave = tid >> 5, lane = tid & 31;
  const int hl = lane >> 4, ln = lane & 15;
  const int row0 = rt * 128 + wave * 16;

  __shared__ __bf16 Wos[64 * 32];

  v8f acc[4] = {};
  for (int k0 = 0; k0 < E_; k0 += 32) {
    __syncthreads();
    {
      const int n = tid >> 2, seg = (tid & 3) * 8;
      *(uint4*)&Wos[n * 32 + seg] =
          *(const uint4*)(Wob + (size_t)(c0 + n) * E_ + k0 + seg);
    }
    __syncthreads();
    v16bf a = load_a_frag(heads + (size_t)row0 * E_ + k0, E_, lane);
#pragma unroll
    for (int ct = 0; ct < 4; ++ct)
      acc[ct] = wmma_bf16(a, load_bT_frag(Wos + ct * 16 * 32, 32, lane), acc[ct]);
  }

#pragma unroll
  for (int ct = 0; ct < 4; ++ct) {
    float bias = bo[c0 + ct * 16 + ln];
#pragma unroll
    for (int r = 0; r < 8; ++r) {
      int row = row0 + hl * 8 + r;
      out[(size_t)row * E_ + c0 + ct * 16 + ln] = acc[ct][r] + bias;
    }
  }
}

// ---------------------------------------------------------------------------
// Launcher
// ---------------------------------------------------------------------------
extern "C" void kernel_launch(void* const* d_in, const int* in_sizes, int n_in,
                              void* d_out, int out_size, void* d_ws, size_t ws_size,
                              hipStream_t stream) {
  (void)in_sizes; (void)n_in; (void)out_size; (void)ws_size;

  const float* X   = (const float*)d_in[0];
  const int*   adj = (const int*)d_in[1];
  const float* Wq  = (const float*)d_in[2];
  const float* Wk  = (const float*)d_in[3];
  const float* Wv  = (const float*)d_in[4];
  const float* Wo  = (const float*)d_in[5];
  const float* bo  = (const float*)d_in[6];
  float* out       = (float*)d_out;

  // Workspace carve (~10 MB)
  const size_t BHND = (size_t)B_ * H_ * N_ * DK_;
  const size_t BNE  = (size_t)B_ * N_ * E_;
  const size_t WSZ  = (size_t)H_ * E_ * DK_;
  __bf16* Qw  = (__bf16*)d_ws;
  __bf16* Kw  = Qw + BHND;
  __bf16* Vtw = Kw + BHND;          // transposed V (B,H,DK,N)
  __bf16* Hw  = Vtw + BHND;
  __bf16* WqT = Hw + BNE;
  __bf16* WkT = WqT + WSZ;
  __bf16* WvT = WkT + WSZ;
  __bf16* Wob = WvT + WSZ;
  unsigned* adjb = (unsigned*)(Wob + (size_t)E_ * E_);

  prep_weights_kernel<<<(4 * 65536) / 256, 256, 0, stream>>>(Wq, Wk, Wv, Wo,
                                                             WqT, WkT, WvT, Wob);
  {
    int total_words = B_ * N_ * N_ / 32;
    adj_to_bits_kernel<<<(total_words + 255) / 256, 256, 0, stream>>>(adj, adjb, total_words);
  }
  qkv_kernel<<<B_ * (N_ / 128) * H_, 256, 0, stream>>>(X, WqT, WkT, WvT, Qw, Kw, Vtw);
  attn_kernel<<<dim3(B_ * H_, N_ / 128), 256, 0, stream>>>(Qw, Kw, Vtw, adjb, Hw);
  proj_kernel<<<dim3(B_ * N_ / 128, E_ / 64), 256, 0, stream>>>(Hw, Wob, bo, out);
}
